// fullPolyModule_38190849196656
// MI455X (gfx1250) — matrix operand order repaired and checked by reading, not compiled, not run on hardware
//
#include <hip/hip_runtime.h>
#include <hip/hip_bf16.h>

typedef __attribute__((ext_vector_type(16))) __bf16 v16bf;
typedef __attribute__((ext_vector_type(8)))  float  v8f;

#define MROWS  (4 * 65536)   // B*L = 262144 rows
#define KDIM   256
#define NDIM   81            // (P+1)^2
#define NT     6             // ceil(81/16) N-tiles (96 padded cols)
#define NKI    (KDIM / 32)   // 8 K-iterations
#define TM     128           // rows per workgroup
#define NTHREADS 256         // 8 wave32 waves

// B-operand staging: [ki][t][lane][16 bf16], hi plane then lo plane
#define BSTAGE_ELEMS (NKI * NT * 32 * 16)          // 24576 bf16 per plane
#define PLANE_BYTES  (BSTAGE_ELEMS * 2)            // 49152 bytes per plane
#define WS_BYTES     (2 * PLANE_BYTES)             // 98304 bytes total
#define SMEM_BYTES   (2 * PLANE_BYTES + TM * 2 * 4)

// Branchless small integer power (e in [0,8])
__device__ __forceinline__ float powi(float x, int e) {
    float r = 1.0f;
    float p = x;
    if (e & 1) r *= p;
    p *= p;
    if (e & 2) r *= p;
    p *= p;
    if (e & 4) r *= p;
    p *= p;
    if (e & 8) r *= p;
    return r;
}

// Decompose one element of W into the WMMA B-operand layout
// (this code's contiguous B order: lane = n%16, K-half by lane>>4, 16 K per lane per 32-K step)
__device__ __forceinline__ void wsplit(const float* __restrict__ W, int idx,
                                       __bf16& h, __bf16& l) {
    const int i  = idx & 15;            // element within lane chunk
    const int ln = (idx >> 4) & 31;     // lane
    const int t  = (idx >> 9) % NT;     // n-tile
    const int ki = idx / (16 * 32 * NT);

    const int n  = t * 16 + (ln & 15);
    const int nc = (n < NDIM) ? n : (NDIM - 1);       // clamp pad columns
    const int k  = ki * 32 + (ln >> 4) * 16 + i;

    const float v = W[(size_t)nc * KDIM + k];
    h = (__bf16)v;
    l = (__bf16)(v - (float)h);
}

// ---------------------------------------------------------------------------
// Prep kernel: stage W hi/lo planes in B-operand layout into workspace.
// ---------------------------------------------------------------------------
__global__ __launch_bounds__(NTHREADS)
void prep_B_kernel(const float* __restrict__ W,
                   __bf16* __restrict__ Bhi,
                   __bf16* __restrict__ Blo)
{
    const int idx = blockIdx.x * NTHREADS + threadIdx.x;   // 0..24575
    if (idx >= BSTAGE_ELEMS) return;
    __bf16 h, l;
    wsplit(W, idx, h, l);
    Bhi[idx] = h;
    Blo[idx] = l;
}

// ---------------------------------------------------------------------------
// Fused kernel: LDS-staged B (async global->LDS) + WMMA bf16 hi/lo GEMM +
// bias/mask + weight store + polynomial monomial dot + wave reduction.
// ---------------------------------------------------------------------------
template <bool USE_WS>
__global__ __launch_bounds__(NTHREADS)
void fullpoly_fused_kernel(const float* __restrict__ x,      // (M, 2)
                           const float* __restrict__ z,      // (M, 256)
                           const float* __restrict__ W,      // (81, 256)
                           const float* __restrict__ b,      // (81)
                           const __bf16* __restrict__ Bhi,   // staged B hi (or null)
                           const __bf16* __restrict__ Blo,   // staged B lo (or null)
                           float* __restrict__ out,          // (M)
                           float* __restrict__ wout)         // (M, 81)
{
    // Single flat LDS block so byte offsets from LDS base are well defined:
    // [0, 48K)  B-hi plane | [48K, 96K) B-lo plane | [96K, 97K) x tile
    __shared__ __align__(16) unsigned char smem[SMEM_BYTES];
    __bf16* sBhi = (__bf16*)(smem);
    __bf16* sBlo = (__bf16*)(smem + PLANE_BYTES);
    float*  sx   = (float*)(smem + 2 * PLANE_BYTES);

    const int tid   = threadIdx.x;
    const int wave  = tid >> 5;       // 0..7
    const int lane  = tid & 31;       // wave32
    const int l15   = lane & 15;
    const int half  = lane >> 4;      // 0 or 1
    const int wgRow = blockIdx.x * TM;

    // ---- Fill LDS: B operand planes + x tile ----
    if (USE_WS) {
        // Async global->LDS DMA copy (no VGPR round-trip), tracked on ASYNCcnt.
        // 3072 x 16B chunks per plane, spread over 256 threads.
        for (int i = tid; i < PLANE_BYTES / 16; i += NTHREADS) {
            const unsigned      ldsH = (unsigned)(i * 16);
            const unsigned      ldsL = (unsigned)(PLANE_BYTES + i * 16);
            const unsigned long gH   = (unsigned long)(const void*)((const char*)Bhi + i * 16);
            const unsigned long gL   = (unsigned long)(const void*)((const char*)Blo + i * 16);
            asm volatile("global_load_async_to_lds_b128 %0, %1, off"
                         :: "v"(ldsH), "v"(gH) : "memory");
            asm volatile("global_load_async_to_lds_b128 %0, %1, off"
                         :: "v"(ldsL), "v"(gL) : "memory");
        }
        asm volatile("s_wait_asynccnt 0x0" ::: "memory");
    } else {
        // One-time in-kernel conversion of W (83 KB, L2-resident)
        for (int idx = tid * 2; idx < BSTAGE_ELEMS; idx += NTHREADS * 2) {
            __bf16 h0, l0, h1, l1;
            wsplit(W, idx, h0, l0);
            wsplit(W, idx + 1, h1, l1);
            sBhi[idx]     = h0;
            sBhi[idx + 1] = h1;
            sBlo[idx]     = l0;
            sBlo[idx + 1] = l1;
        }
    }
    sx[tid] = x[(size_t)wgRow * 2 + tid];
    __syncthreads();

    // A-matrix row handled by this lane (lanes 0-15 / 16-31 both map M = lane%16)
    const int    m    = wgRow + wave * 16 + l15;
    const float* zrow = z + (size_t)m * KDIM;

    v8f acc[NT] = {};

    for (int ki = 0; ki < NKI; ++ki) {
        const int kb = ki * 32;
        // Prefetch next z chunk for this row into near caches
        if (ki + 1 < NKI)
            __builtin_prefetch(zrow + kb + 32, 0, 3);

        // ---- A tile: bf16 hi/lo split of z[m][kb..kb+31], contiguous k order per lane half ----
        // elems 0..15 <- K = kb + half*16 + 0..15   (contiguous, matches B staging)
        float af[16];
        {
            const float* p0 = zrow + kb + half * 16;
            *(float4*)&af[0]  = *(const float4*)(p0);
            *(float4*)&af[4]  = *(const float4*)(p0 + 4);
            *(float4*)&af[8]  = *(const float4*)(p0 + 8);
            *(float4*)&af[12] = *(const float4*)(p0 + 12);
        }
        v16bf ahi, alo;
#pragma unroll
        for (int i = 0; i < 16; ++i) {
            __bf16 h = (__bf16)af[i];
            ahi[i] = h;
            alo[i] = (__bf16)(af[i] - (float)h);
        }

        // ---- B tiles from LDS (two aligned 32B ds reads each) ----
#pragma unroll
        for (int t = 0; t < NT; ++t) {
            const int off = ((ki * NT + t) * 32 + lane) * 16;
            const v16bf bhi = *(const v16bf*)(sBhi + off);
            const v16bf blo = *(const v16bf*)(sBlo + off);

            // fp32-accurate product via hi/lo decomposition (lo*lo negligible)
            acc[t] = __builtin_amdgcn_wmma_f32_16x16x32_bf16(
                         false, ahi, false, bhi, (short)0, acc[t], false, false);
            acc[t] = __builtin_amdgcn_wmma_f32_16x16x32_bf16(
                         false, ahi, false, blo, (short)0, acc[t], false, false);
            acc[t] = __builtin_amdgcn_wmma_f32_16x16x32_bf16(
                         false, alo, false, bhi, (short)0, acc[t], false, false);
        }
    }

    // ---- Epilogue: bias, mask, store weights, polynomial dot ----
    // C/D layout: lane l, VGPR v -> row = v + 8*half, col n = l15
    float dot[8] = {0.f, 0.f, 0.f, 0.f, 0.f, 0.f, 0.f, 0.f};

#pragma unroll
    for (int t = 0; t < NT; ++t) {
        const int n = t * 16 + l15;
        if (n < NDIM) {
            const float bias = b[n];
            const float msk  = (n == 0 || n == 1 || n == 9) ? 0.0f : 1.0f;
            const int   p    = n / 9;       // x0 exponent
            const int   kk   = n % 9;       // x1 exponent
#pragma unroll
            for (int v = 0; v < 8; ++v) {
                const int lr = wave * 16 + v + 8 * half;   // local row in tile
                const int r  = wgRow + lr;                 // global row
                const float wgt = (acc[t][v] + bias) * msk;
                wout[(size_t)r * NDIM + n] = wgt;
                const float x0 = sx[lr * 2 + 0];
                const float x1 = sx[lr * 2 + 1];
                dot[v] += wgt * powi(x0, p) * powi(x1, kk);
            }
        }
    }

    // Reduce dot across the 16 lanes of each half (masks 1,2,4,8 stay in-group)
#pragma unroll
    for (int v = 0; v < 8; ++v) {
        float s = dot[v];
        s += __shfl_xor(s, 1, 32);
        s += __shfl_xor(s, 2, 32);
        s += __shfl_xor(s, 4, 32);
        s += __shfl_xor(s, 8, 32);
        if (l15 == 0) {
            const int r = wgRow + wave * 16 + v + 8 * half;
            out[r] = s;
        }
    }
}

extern "C" void kernel_launch(void* const* d_in, const int* in_sizes, int n_in,
                              void* d_out, int out_size, void* d_ws, size_t ws_size,
                              hipStream_t stream) {
    const float* x = (const float*)d_in[0];   // (4, 65536, 2)
    const float* z = (const float*)d_in[1];   // (4, 65536, 256)
    const float* W = (const float*)d_in[2];   // (81, 256)
    const float* b = (const float*)d_in[3];   // (81,)

    float* out  = (float*)d_out;              // first output: (4, 65536, 1) flat
    float* wout = out + (size_t)MROWS;        // second output: (4, 65536, 9, 9) flat

    const dim3 grid(MROWS / TM);              // 2048 workgroups

    if (d_ws != nullptr && ws_size >= (size_t)WS_BYTES) {
        __bf16* Bhi = (__bf16*)d_ws;
        __bf16* Blo = Bhi + BSTAGE_ELEMS;
        const int prepBlocks = (BSTAGE_ELEMS + NTHREADS - 1) / NTHREADS;  // 96
        prep_B_kernel<<<prepBlocks, NTHREADS, 0, stream>>>(W, Bhi, Blo);
        fullpoly_fused_kernel<true><<<grid, NTHREADS, 0, stream>>>(
            x, z, W, b, Bhi, Blo, out, wout);
    } else {
        fullpoly_fused_kernel<false><<<grid, NTHREADS, 0, stream>>>(
            x, z, W, b, nullptr, nullptr, out, wout);
    }
}
